// AttnBlock_79517024518715
// MI455X (gfx1250) — matrix operand order repaired and checked by reading, not compiled, not run on hardware
//
#include <hip/hip_runtime.h>

#define DEVINL __device__ __forceinline__

typedef __attribute__((ext_vector_type(16))) __bf16 v16bf;
typedef __attribute__((ext_vector_type(8)))  __bf16 v8bf;
typedef __attribute__((ext_vector_type(8)))  float  v8f;

// Problem constants
constexpr int B_   = 64;
constexpr int T_   = 256;
constexpr int CIN  = 512;
constexpr int COUT = 1024;
constexpr int CH   = 512;   // C_OUT/2 == ATTN_CH
constexpr int F_   = 512;   // ATTN_CH
constexpr int S_   = 196;   // H*W
constexpr int SP   = 224;   // S padded to multiple of 32
constexpr int TP   = 258;   // T + 2 (left zero pad rows)

// ---------------- helpers ----------------

DEVINL unsigned short f2bfu(float f) {
  unsigned u = __builtin_bit_cast(unsigned, f);
  unsigned r = u + 0x7FFFu + ((u >> 16) & 1u);   // round-to-nearest-even
  return (unsigned short)(r >> 16);
}

DEVINL v8f wmma_bf16(v16bf a, v16bf b, v8f c) {
  // D = A(16x32) * B(32x16) + C, fp32 accumulate
  return __builtin_amdgcn_wmma_f32_16x16x32_bf16(
      /*neg_a=*/false, a, /*neg_b=*/false, b,
      /*c_mod=*/(short)0, c, /*reuse_a=*/false, /*reuse_b=*/false);
}

// A fragment: 16x32 bf16, row m = lane&15, K interleaved: lanes<16 hold
// K in {0..7, 16..23}, lanes>=16 hold K in {8..15, 24..31}.
DEVINL v16bf load_a(const unsigned short* rowbase, int ld, int lane) {
  const __bf16* p = (const __bf16*)(rowbase + (size_t)(lane & 15) * ld + ((lane >> 4) << 3));
  v8bf lo = *(const v8bf*)p;
  v8bf hi = *(const v8bf*)(p + 16);
  v16bf r;
#pragma unroll
  for (int i = 0; i < 8; ++i) { r[i] = lo[i]; r[i + 8] = hi[i]; }
  return r;
}

// B fragment: 32x16 bf16, column n = lane&15, K interleaved: lanes<16 hold
// K in {0..7, 16..23}, lanes>=16 hold K in {8..15, 24..31}.
DEVINL v16bf load_b(const unsigned short* colbase, int ld, int lane) {
  const __bf16* p = (const __bf16*)(colbase + (size_t)(lane & 15) * ld + ((lane >> 4) << 3));
  v8bf lo = *(const v8bf*)p;
  v8bf hi = *(const v8bf*)(p + 16);
  v16bf r;
#pragma unroll
  for (int i = 0; i < 8; ++i) { r[i] = lo[i]; r[i + 8] = hi[i]; }
  return r;
}

// ---------------- prep kernels ----------------

// Weight-norm: wt[j][co][ci] = g[co] * v[co][ci][j] * rsqrt(sum v^2)
__global__ void __launch_bounds__(256) k_wnorm(const float* __restrict__ v,
                                               const float* __restrict__ g,
                                               unsigned short* __restrict__ wt) {
  int co = blockIdx.x;
  int tid = threadIdx.x;
  __shared__ float red[256];
  float s = 0.f;
  for (int e = tid; e < CIN * 3; e += 256) {
    float x = v[(size_t)co * CIN * 3 + e];
    s += x * x;
  }
  red[tid] = s;
  __syncthreads();
  for (int off = 128; off > 0; off >>= 1) {
    if (tid < off) red[tid] += red[tid + off];
    __syncthreads();
  }
  float scale = g[co] * rsqrtf(red[0]);
  for (int e = tid; e < CIN * 3; e += 256) {
    int ci = e / 3, j = e % 3;
    wt[((size_t)j * COUT + co) * CIN + ci] = f2bfu(v[(size_t)co * CIN * 3 + e] * scale);
  }
}

__global__ void __launch_bounds__(256) k_cvt(const float* __restrict__ src,
                                             unsigned short* __restrict__ dst, int n) {
  for (int i = blockIdx.x * blockDim.x + threadIdx.x; i < n; i += gridDim.x * blockDim.x)
    dst[i] = f2bfu(src[i]);
}

// xt[b][r][ci] bf16, r in [0,258): rows 0,1 zero, row r>=2 = x[b][ci][r-2]
__global__ void __launch_bounds__(256) k_xt(const float* __restrict__ x,
                                            unsigned short* __restrict__ xt) {
  int total = B_ * TP * CIN;
  for (int idx = blockIdx.x * blockDim.x + threadIdx.x; idx < total;
       idx += gridDim.x * blockDim.x) {
    int ci = idx % CIN;
    int r  = (idx / CIN) % TP;
    int b  = idx / (CIN * TP);
    float v = (r < 2) ? 0.f : x[((size_t)b * CIN + ci) * T_ + (r - 2)];
    xt[idx] = f2bfu(v);
  }
}

// featT[b][p][f] and featC[b][f][p] bf16 (p padded to 224 with zeros)
__global__ void __launch_bounds__(256) k_feat(const float* __restrict__ img,
                                              unsigned short* __restrict__ featT,
                                              unsigned short* __restrict__ featC) {
  int total = B_ * SP * F_;
  for (int idx = blockIdx.x * blockDim.x + threadIdx.x; idx < total;
       idx += gridDim.x * blockDim.x) {
    int f = idx % F_;
    int p = (idx / F_) % SP;
    int b = idx / (F_ * SP);
    float v = (p < S_) ? img[((size_t)b * F_ + f) * S_ + p] : 0.f;
    unsigned short bf = f2bfu(v);
    featT[idx] = bf;
    featC[((size_t)b * F_ + f) * SP + p] = bf;
  }
}

// ---------------- conv + GLU ----------------
// One wave per 16(co in first half) x 16(t) output tile; accumulates both
// halves (a and gate b) over 3 taps x 16 K-chunks => 96 WMMAs / wave.
__global__ void __launch_bounds__(256) k_conv(const unsigned short* __restrict__ wt,
                                              const unsigned short* __restrict__ xt,
                                              const float* __restrict__ convb,
                                              float* __restrict__ yglu,
                                              unsigned short* __restrict__ yt) {
  int lane = threadIdx.x & 31;
  int wid  = threadIdx.x >> 5;
  int tile = blockIdx.x * 8 + wid;          // 0 .. 32767
  int b    = tile >> 9;
  int r    = tile & 511;
  int co0  = (r >> 4) << 4;                 // 0..496 step 16 (first half channels)
  int t0   = (r & 15) << 4;

  v8f accA = {0.f, 0.f, 0.f, 0.f, 0.f, 0.f, 0.f, 0.f};
  v8f accB = accA;
  const unsigned short* xbase = xt + ((size_t)b * TP + t0) * CIN;
#pragma unroll
  for (int j = 0; j < 3; ++j) {
    const unsigned short* wa = wt + ((size_t)j * COUT + co0) * CIN;
    const unsigned short* wb = wt + ((size_t)j * COUT + co0 + CH) * CIN;
    const unsigned short* xb = xbase + (size_t)j * CIN;  // tap shifts the row
    for (int kc = 0; kc < CIN; kc += 32) {
      v16bf bm = load_b(xb + kc, CIN, lane);
      v16bf a1 = load_a(wa + kc, CIN, lane);
      accA = wmma_bf16(a1, bm, accA);
      v16bf a2 = load_a(wb + kc, CIN, lane);
      accB = wmma_bf16(a2, bm, accB);
    }
  }
  int n = lane & 15, hi = lane >> 4;
  int t = t0 + n;
  int cbase = co0 + hi * 8;                  // D: m = r + 8*hi -> contiguous c
  float*          yg = yglu + ((size_t)b * T_ + t) * CH + cbase;
  unsigned short* yo = yt   + ((size_t)b * T_ + t) * CH + cbase;
#pragma unroll
  for (int i = 0; i < 8; ++i) {
    float av = accA[i] + convb[cbase + i];
    float bv = accB[i] + convb[cbase + i + CH];
    float gv = av / (1.f + __expf(-bv));     // GLU
    yg[i] = gv;
    yo[i] = f2bfu(gv);
  }
}

// ---------------- fused fc1 -> attention -> fc2 -> residual ----------------
// One block (8 waves) per (b, 16-row t tile). All intermediates in LDS.
__global__ void __launch_bounds__(256) k_attn(const unsigned short* __restrict__ yt,
                                              const unsigned short* __restrict__ fc1w,
                                              const float* __restrict__ fc1b,
                                              const float* __restrict__ we,
                                              const unsigned short* __restrict__ featT,
                                              const unsigned short* __restrict__ featC,
                                              const unsigned short* __restrict__ fc2w,
                                              const float* __restrict__ fc2b,
                                              const float* __restrict__ yglu,
                                              const float* __restrict__ x,
                                              float* __restrict__ out0,
                                              float* __restrict__ out3) {
  __shared__ __attribute__((aligned(16))) unsigned short sh_h[16 * F_];  // h bf16
  __shared__ __attribute__((aligned(16))) float          sh_s[16 * SP];  // scores f32
  __shared__ __attribute__((aligned(16))) unsigned short sh_p[16 * SP];  // probs bf16
  __shared__ __attribute__((aligned(16))) unsigned short sh_c[16 * F_];  // ctx bf16
  __shared__ float red[256];
  __shared__ float rowv[16];

  int tid = threadIdx.x, lane = tid & 31, wid = tid >> 5;
  int b  = blockIdx.x >> 4;
  int t0 = (blockIdx.x & 15) << 4;
  int n = lane & 15, hi = lane >> 4;

  const unsigned short* ybase = yt + ((size_t)b * T_ + t0) * CH;

  // ---- fc1: h[t,f] = sum_c y[t,c]*W1[f,c] + b1[f] + word_embed ----
  for (int ft = wid; ft < F_ / 16; ft += 8) {
    v8f acc = {0.f, 0.f, 0.f, 0.f, 0.f, 0.f, 0.f, 0.f};
    const unsigned short* arow = fc1w + (size_t)ft * 16 * CH;
    for (int kc = 0; kc < CH; kc += 32) {
      v16bf a  = load_a(arow + kc, CH, lane);
      v16bf bm = load_b(ybase + kc, CH, lane);
      acc = wmma_bf16(a, bm, acc);
    }
    int fb = ft * 16 + hi * 8;
    const float* wrow = we + ((size_t)b * T_ + (t0 + n)) * F_ + fb;
    unsigned short* hrow = sh_h + n * F_ + fb;
#pragma unroll
    for (int i = 0; i < 8; ++i)
      hrow[i] = f2bfu(acc[i] + fc1b[fb + i] + wrow[i]);
  }
  __syncthreads();

  // ---- scores: s[t,p] = sum_f featT[p,f]*h[t,f] ----
  for (int pt = wid; pt < SP / 16; pt += 8) {
    v8f acc = {0.f, 0.f, 0.f, 0.f, 0.f, 0.f, 0.f, 0.f};
    const unsigned short* arow = featT + ((size_t)b * SP + pt * 16) * F_;
    for (int kc = 0; kc < F_; kc += 32) {
      v16bf a  = load_a(arow + kc, F_, lane);
      v16bf bm = load_b(sh_h + kc, F_, lane);
      acc = wmma_bf16(a, bm, acc);
    }
    float* srow = sh_s + n * SP + pt * 16 + hi * 8;
#pragma unroll
    for (int i = 0; i < 8; ++i) srow[i] = acc[i];
  }
  __syncthreads();

  // mask padded spatial positions
  for (int idx = tid; idx < 16 * (SP - S_); idx += 256) {
    int rr = idx / (SP - S_);
    int p  = S_ + idx % (SP - S_);
    sh_s[rr * SP + p] = -1e30f;
  }
  __syncthreads();

  // ---- softmax over p (16 threads per row) ----
  int r = tid >> 4, sub = tid & 15;
  float m = -1e30f;
  for (int p = sub; p < SP; p += 16) m = fmaxf(m, sh_s[r * SP + p]);
  red[tid] = m;
  __syncthreads();
  if (sub == 0) {
    float mm = red[r * 16];
    for (int i = 1; i < 16; ++i) mm = fmaxf(mm, red[r * 16 + i]);
    rowv[r] = mm;
  }
  __syncthreads();
  float rm = rowv[r], s = 0.f;
  for (int p = sub; p < SP; p += 16) {
    float e = __expf(sh_s[r * SP + p] - rm);
    sh_s[r * SP + p] = e;
    s += e;
  }
  red[tid] = s;
  __syncthreads();
  if (sub == 0) {
    float ss = 0.f;
    for (int i = 0; i < 16; ++i) ss += red[r * 16 + i];
    rowv[r] = 1.f / ss;
  }
  __syncthreads();
  float inv = rowv[r];
  for (int p = sub; p < SP; p += 16) {
    float pr = sh_s[r * SP + p] * inv;
    bool valid = (p < S_);
    sh_p[r * SP + p] = f2bfu(valid ? pr : 0.f);
    if (valid) out3[((size_t)b * T_ + (t0 + r)) * S_ + p] = pr;
  }
  __syncthreads();

  // ---- ctx: c[t,f] = sum_p featC[f,p]*prob[t,p] ----
  for (int ft = wid; ft < F_ / 16; ft += 8) {
    v8f acc = {0.f, 0.f, 0.f, 0.f, 0.f, 0.f, 0.f, 0.f};
    const unsigned short* arow = featC + ((size_t)b * F_ + ft * 16) * SP;
    for (int kc = 0; kc < SP; kc += 32) {
      v16bf a  = load_a(arow + kc, SP, lane);
      v16bf bm = load_b(sh_p + kc, SP, lane);
      acc = wmma_bf16(a, bm, acc);
    }
    int fb = ft * 16 + hi * 8;
    unsigned short* crow = sh_c + n * F_ + fb;
#pragma unroll
    for (int i = 0; i < 8; ++i) crow[i] = f2bfu(acc[i]);
  }
  __syncthreads();

  // ---- fc2 + bias + identity(GLU) + residual(x), write (B, C, T) ----
  for (int ct = wid; ct < CH / 16; ct += 8) {
    v8f acc = {0.f, 0.f, 0.f, 0.f, 0.f, 0.f, 0.f, 0.f};
    const unsigned short* arow = fc2w + (size_t)ct * 16 * F_;
    for (int kc = 0; kc < F_; kc += 32) {
      v16bf a  = load_a(arow + kc, F_, lane);
      v16bf bm = load_b(sh_c + kc, F_, lane);
      acc = wmma_bf16(a, bm, acc);
    }
    int t  = t0 + n;
    int cb = ct * 16 + hi * 8;
    const float* yg = yglu + ((size_t)b * T_ + t) * CH + cb;
#pragma unroll
    for (int i = 0; i < 8; ++i) {
      int c = cb + i;
      float v = acc[i] + fc2b[c] + yg[i] + x[((size_t)b * CIN + c) * T_ + t];
      out0[((size_t)b * CH + c) * T_ + t] = v;
    }
  }
}

// ---------------- host launcher ----------------

extern "C" void kernel_launch(void* const* d_in, const int* in_sizes, int n_in,
                              void* d_out, int out_size, void* d_ws, size_t ws_size,
                              hipStream_t stream) {
  (void)in_sizes; (void)n_in; (void)out_size; (void)ws_size;

  const float* x     = (const float*)d_in[0];
  const float* wemb  = (const float*)d_in[1];
  const float* img   = (const float*)d_in[2];
  // d_in[3] = prev_attn (unused by the reference math)
  const float* convv = (const float*)d_in[4];
  const float* convg = (const float*)d_in[5];
  const float* convb = (const float*)d_in[6];
  const float* fc1wf = (const float*)d_in[7];
  const float* fc1b  = (const float*)d_in[8];
  const float* fc2wf = (const float*)d_in[9];
  const float* fc2b  = (const float*)d_in[10];

  float* out0 = (float*)d_out;                         // (B, 512, 256)
  float* out1 = out0 + (size_t)B_ * CH * T_;           // word_embed passthrough
  float* out2 = out1 + (size_t)B_ * T_ * F_;           // img_conv passthrough
  float* out3 = out2 + (size_t)B_ * F_ * S_;           // score (B, T, 196)

  char* w = (char*)d_ws;
  auto alloc = [&](size_t bytes) -> char* {
    char* p = w;
    w += (bytes + 255) & ~(size_t)255;
    return p;
  };
  unsigned short* wt    = (unsigned short*)alloc((size_t)3 * COUT * CIN * 2);
  unsigned short* fc1wb = (unsigned short*)alloc((size_t)F_ * CH * 2);
  unsigned short* fc2wb = (unsigned short*)alloc((size_t)CH * F_ * 2);
  unsigned short* xt    = (unsigned short*)alloc((size_t)B_ * TP * CIN * 2);
  unsigned short* featT = (unsigned short*)alloc((size_t)B_ * SP * F_ * 2);
  unsigned short* featC = (unsigned short*)alloc((size_t)B_ * F_ * SP * 2);
  unsigned short* yt    = (unsigned short*)alloc((size_t)B_ * T_ * CH * 2);
  float*          yglu  = (float*)alloc((size_t)B_ * T_ * CH * 4);

  // --- stage 0: weight prep + bf16 staging ---
  k_wnorm<<<COUT, 256, 0, stream>>>(convv, convg, wt);
  k_cvt<<<512, 256, 0, stream>>>(fc1wf, fc1wb, F_ * CH);
  k_cvt<<<512, 256, 0, stream>>>(fc2wf, fc2wb, CH * F_);
  k_xt<<<2048, 256, 0, stream>>>(x, xt);
  k_feat<<<2048, 256, 0, stream>>>(img, featT, featC);

  // passthrough outputs
  hipMemcpyAsync(out1, wemb, (size_t)B_ * T_ * F_ * sizeof(float),
                 hipMemcpyDeviceToDevice, stream);
  hipMemcpyAsync(out2, img, (size_t)B_ * F_ * S_ * sizeof(float),
                 hipMemcpyDeviceToDevice, stream);

  // --- stage 1: weight-normed causal conv + GLU (WMMA) ---
  k_conv<<<4096, 256, 0, stream>>>(wt, xt, convb, yglu, yt);

  // --- stage 2: fused fc1 -> attention -> fc2 -> residual (WMMA) ---
  k_attn<<<B_ * (T_ / 16), 256, 0, stream>>>(yt, fc1wb, fc1b, wemb, featT, featC,
                                             fc2wb, fc2b, yglu, x, out0, out3);
}
